// MQFLayerWrapper_31593779429381
// MI455X (gfx1250) — compile-verified, matrix operand order checked by reading, not run on hardware
//
#include <hip/hip_runtime.h>
#include <hip/hip_bf16.h>
#include <limits.h>

typedef int v8i __attribute__((ext_vector_type(8)));

#define BATCH 16
#define CIN   128
#define COUT  256
#define HH    56
#define WW    56
#define HW    (HH*WW)        // 3136
#define MTOT  (BATCH*HW)     // 50176
#define KTOT  (9*CIN)        // 1152
#define KSTEPS (KTOT/64)     // 18

// workspace slot layout (int/float aliased slots)
#define WS_MINKEY 0
#define WS_MAXKEY 1
#define WS_ISCALE 2
#define WS_ZP     3
#define WS_ZPI    4
#define WS_WSMAX0 5
#define WS_WSMAX1 6
#define WS_ACCMAX0 7
#define WS_ACCMAX1 8
#define WS_OSCALE 9
#define WS_WSCALE 16        // float[256]
#define WS_BINT   272       // int[256]
#define WS_TSUM   528       // int[256*9]
#define WS_MFIXED 3072      // float[256]
#define WS_XQ_OFF 16384                         // bytes: int8 x, NHWC, 6,422,528 B
#define WS_W8_OFF (16384 + BATCH*CIN*HW)        // bytes: pre-swizzled int8 weights, 589,824 B, half used

__device__ __forceinline__ int f2key(float f) {
  int i = __float_as_int(f);
  return i >= 0 ? i : (i ^ 0x7fffffff);
}
__device__ __forceinline__ float key2f(int k) {
  return __int_as_float(k >= 0 ? k : (k ^ 0x7fffffff));
}

__global__ void k_init(int* wsi) {
  if (threadIdx.x == 0) {
    wsi[WS_MINKEY] = INT_MAX;
    wsi[WS_MAXKEY] = INT_MIN;
    wsi[WS_WSMAX0] = 0;
    wsi[WS_WSMAX1] = 0;
    wsi[WS_ACCMAX0] = 0;
    wsi[WS_ACCMAX1] = 0;
  }
}

__global__ void k_minmax(const float4* __restrict__ x, int n4, int* wsi) {
  __shared__ float smin[256], smax[256];
  int tid = threadIdx.x;
  float lmin = 3.4e38f, lmax = -3.4e38f;
  for (int i = blockIdx.x * blockDim.x + tid; i < n4; i += gridDim.x * blockDim.x) {
    float4 v = x[i];
    lmin = fminf(lmin, fminf(fminf(v.x, v.y), fminf(v.z, v.w)));
    lmax = fmaxf(lmax, fmaxf(fmaxf(v.x, v.y), fmaxf(v.z, v.w)));
  }
  smin[tid] = lmin; smax[tid] = lmax;
  __syncthreads();
  for (int s = 128; s > 0; s >>= 1) {
    if (tid < s) {
      smin[tid] = fminf(smin[tid], smin[tid + s]);
      smax[tid] = fmaxf(smax[tid], smax[tid + s]);
    }
    __syncthreads();
  }
  if (tid == 0) {
    atomicMin(&wsi[WS_MINKEY], f2key(smin[0]));
    atomicMax(&wsi[WS_MAXKEY], f2key(smax[0]));
  }
}

__global__ void k_scale(int* wsi, float* wsf) {
  float xmin = key2f(wsi[WS_MINKEY]);
  float xmax = key2f(wsi[WS_MAXKEY]);
  float iscale = (xmax > xmin) ? (xmax - xmin) / 255.f : 1e-9f;
  float zp = rintf(-128.f - xmin / iscale);
  wsf[WS_ISCALE] = iscale;
  wsf[WS_ZP] = zp;
  wsi[WS_ZPI] = (int)zp;
}

// per-output-channel weight quantization; writes weights pre-swizzled into the
// exact 8-bit B-fragment (64x16 K x N) VGPR layout so the GEMM loads them
// as two contiguous b128 per lane.
__global__ __launch_bounds__(128) void k_wquant(const float* __restrict__ w,
                                                const float* __restrict__ bias,
                                                int* wsi, float* wsf,
                                                signed char* w8) {
  int cout = blockIdx.x;
  int tid = threadIdx.x;  // 0..127 == ci
  __shared__ float red[128];
  __shared__ float s_scale;
  __shared__ int tsum[9];
  const float* wrow = w + (size_t)cout * CIN * 9 + (size_t)tid * 9;
  float wv[9];
  float lmax = 0.f;
#pragma unroll
  for (int t = 0; t < 9; ++t) { wv[t] = wrow[t]; lmax = fmaxf(lmax, fabsf(wv[t])); }
  red[tid] = lmax;
  __syncthreads();
  for (int s = 64; s > 0; s >>= 1) {
    if (tid < s) red[tid] = fmaxf(red[tid], red[tid + s]);
    __syncthreads();
  }
  bool g1 = (cout >= 128);
  float denom = g1 ? 255.f : 15.f;
  float qmin = g1 ? -128.f : -8.f;
  float qmax = g1 ? 127.f : 7.f;
  if (tid == 0) {
    float sc = fmaxf(red[0] * 2.f / denom, 1e-9f);
    s_scale = sc;
    wsf[WS_WSCALE + cout] = sc;
    float iscale = wsf[WS_ISCALE];
    wsi[WS_BINT + cout] = (int)rintf(bias[cout] / (iscale * sc));
    atomicMax(&wsi[g1 ? WS_WSMAX1 : WS_WSMAX0], __float_as_int(sc));
  }
  if (tid < 9) tsum[tid] = 0;
  __syncthreads();
  float sc = s_scale;
#pragma unroll
  for (int t = 0; t < 9; ++t) {
    float q = rintf(wv[t] / sc);
    q = fminf(fmaxf(q, qmin), qmax);
    int qi = (int)q;
    atomicAdd(&tsum[t], qi);
    // K index: tap-major, ci within
    int k = t * CIN + tid;
    int kc = k >> 6;
    int krem = k & 63;
    int bsel = krem & 3;
    int v = ((krem >> 5) & 1) * 4 + ((krem >> 4) & 1) * 2 + ((krem >> 2) & 1);
    int half = (krem >> 3) & 1;
    int lane = half * 16 + (cout & 15);
    int ng = cout >> 4;
    w8[(size_t)(kc * 16 + ng) * 2048 + (size_t)lane * 32 + v * 4 + bsel] = (signed char)qi;
  }
  __syncthreads();
  if (tid < 9) wsi[WS_TSUM + cout * 9 + tid] = tsum[tid];
}

// NCHW f32 -> NHWC int8 activation quantization, LDS-transposed so both the
// global read and the int8 write are coalesced.
__global__ __launch_bounds__(256) void k_xquant(const float* __restrict__ x,
                                                const float* __restrict__ wsf,
                                                signed char* __restrict__ xq) {
  __shared__ float tile[32][33];
  int bb = blockIdx.z;
  int cib = blockIdx.y * 32;
  int rmb = blockIdx.x * 32;
  int tx = threadIdx.x & 31;
  int ty = threadIdx.x >> 5;  // 0..7
  float iscale = wsf[WS_ISCALE];
  float zp = wsf[WS_ZP];
#pragma unroll
  for (int r = 0; r < 32; r += 8) {
    int ci = cib + ty + r;
    tile[ty + r][tx] = x[((size_t)(bb * CIN + ci)) * HW + rmb + tx];
  }
  __syncthreads();
#pragma unroll
  for (int r = 0; r < 32; r += 8) {
    int rm = rmb + ty + r;
    float v = tile[tx][ty + r];
    float q = fminf(fmaxf(rintf(v / iscale + zp), -128.f), 127.f);
    xq[((size_t)(bb * HW + rm)) * CIN + cib + tx] = (signed char)q;
  }
}

// int8 implicit-GEMM conv: M=50176 pixels, N=256 couts, K=1152 (9 taps x 128 ci)
// block = 8 waves, 64x64 output tile, fully unrolled K loop, ping-pong LDS A
// (one barrier per k-step), software-pipelined A/B global fetch.
__global__ __launch_bounds__(256) void k_conv(const signed char* __restrict__ xq,
                                              const signed char* __restrict__ w8,
                                              const int* __restrict__ wsi,
                                              int* __restrict__ out,
                                              int* __restrict__ wsAcc) {
  __shared__ signed char Asmem[2][64 * 64];
  __shared__ int Ts[64 * 9];
  __shared__ int bints[64];
  int tid = threadIdx.x;
  int mbase = blockIdx.x * 64;
  int nbase = blockIdx.y * 64;
  for (int i = tid; i < 576; i += 256)
    Ts[i] = wsi[WS_TSUM + (nbase + i / 9) * 9 + (i % 9)];
  if (tid < 64) bints[tid] = wsi[WS_BINT + nbase + tid];
  int zpi = wsi[WS_ZPI];

  int lane = tid & 31;
  int wv = tid >> 5;        // wave id 0..7 (wave32)
  int msub = wv & 3;        // M sub-tile 0..3
  int npb = (wv >> 2) * 2;  // N sub-tile pair base {0,2}
  int half = lane >> 4;

  v8i acc0 = {0, 0, 0, 0, 0, 0, 0, 0};
  v8i acc1 = acc0;

  // cooperative A-tile gather setup: thread -> (row, 16B segment); border
  // validity per tap dimension precomputed once.
  int arow = tid >> 2;
  int seg = tid & 3;
  int m = mbase + arow;
  int bb = m / HW;
  int rem = m - bb * HW;
  int y0 = rem / WW;
  int x0 = rem - y0 * WW;
  bool vy[3] = {y0 > 0, true, y0 < HH - 1};
  bool vx[3] = {x0 > 0, true, x0 < WW - 1};
  const signed char* src0 = xq + ((size_t)((bb * HH + y0) * WW + x0)) * CIN + seg * 16;

  auto fetchA = [&](int kc) -> int4 {
    int tap = kc >> 1;
    int dy = tap / 3;       // 0..2 (offset dy-1)
    int dxn = tap - dy * 3; // 0..2 (offset dxn-1)
    int4 v = make_int4(0, 0, 0, 0);
    if (vy[dy] && vx[dxn])
      v = *(const int4*)(src0 + ((dy - 1) * WW + (dxn - 1)) * CIN + (kc & 1) * 64);
    return v;
  };

  const signed char* wbase =
      w8 + (size_t)(blockIdx.y * 4 + npb) * 2048 + (size_t)lane * 32;
  auto fetchB = [&](int kc, int t) -> v8i {
    const int4* p = (const int4*)(wbase + (size_t)kc * (16 * 2048) + t * 2048);
    int4 a0 = p[0], a1 = p[1];
    v8i r = {a0.x, a0.y, a0.z, a0.w, a1.x, a1.y, a1.z, a1.w};
    return r;
  };

  // software pipeline prologue
  int4 aNext = fetchA(0);
  v8i b0n = fetchB(0, 0);
  v8i b1n = fetchB(0, 1);
  const int rowoff = (msub * 16 + (lane & 15)) * 64;

#pragma unroll
  for (int kc = 0; kc < KSTEPS; ++kc) {
    // store this step's A tile (written buffer was last read two steps ago;
    // the intervening barriers make the store safe with one barrier per step)
    *(int4*)&Asmem[kc & 1][arow * 64 + seg * 16] = aNext;
    __syncthreads();

    v8i bf0 = b0n, bf1 = b1n;
    if (kc + 1 < KSTEPS) {
      aNext = fetchA(kc + 1);
      b0n = fetchB(kc + 1, 0);
      b1n = fetchB(kc + 1, 1);
    }

    // A fragment per the ISA 8-bit 16x64 A layout: 4x ds_load_b64
    const signed char* arowp = &Asmem[kc & 1][rowoff];
    v8i a;
#pragma unroll
    for (int j = 0; j < 4; ++j) {
      unsigned long long d = *(const unsigned long long*)(arowp + j * 16 + half * 8);
      a[2 * j] = (int)(unsigned int)(d & 0xffffffffull);
      a[2 * j + 1] = (int)(unsigned int)(d >> 32);
    }

    acc0 = __builtin_amdgcn_wmma_i32_16x16x64_iu8(true, a, true, bf0, acc0, false, false);
    acc1 = __builtin_amdgcn_wmma_i32_16x16x64_iu8(true, a, true, bf1, acc1, false, false);
  }

  // epilogue: zp correction (partial-tap sums) + bias, store raw i32, track |acc| max
  int amax = 0;
  int nloc = lane & 15;
#pragma unroll
  for (int t = 0; t < 2; ++t) {
    int nsub = npb + t;
    int cl = nsub * 16 + nloc;
    int cout = nbase + cl;
    v8i acc = t ? acc1 : acc0;
#pragma unroll
    for (int r = 0; r < 8; ++r) {
      int ml = r + 8 * half;
      int mm = mbase + msub * 16 + ml;
      int b2 = mm / HW;
      int rm = mm - b2 * HW;
      int yy = rm / WW;
      int xx = rm - yy * WW;
      int S = 0;
#pragma unroll
      for (int ddy = 0; ddy < 3; ++ddy)
#pragma unroll
        for (int ddx = 0; ddx < 3; ++ddx) {
          int yv = yy + ddy - 1, xv = xx + ddx - 1;
          if (yv >= 0 && yv < HH && xv >= 0 && xv < WW) S += Ts[cl * 9 + ddy * 3 + ddx];
        }
      int av = acc[r] - zpi * S + bints[cl];
      out[((size_t)(b2 * COUT + cout)) * HW + yy * WW + xx] = av;
      int aa = av < 0 ? -av : av;
      amax = aa > amax ? aa : amax;
    }
  }
  atomicMax(&wsAcc[blockIdx.y < 2 ? 0 : 1], amax);
}

__global__ void k_outscale(int* wsi, float* wsf) {
  float iscale = wsf[WS_ISCALE];
  float g0 = (float)wsi[WS_ACCMAX0] * iscale * __int_as_float(wsi[WS_WSMAX0]);
  float g1 = (float)wsi[WS_ACCMAX1] * iscale * __int_as_float(wsi[WS_WSMAX1]);
  float gmax = fmaxf(g0, g1);
  wsf[WS_OSCALE] = (gmax > 1e-6f) ? gmax / 127.f : iscale;
}

__global__ void k_mfixed(float* wsf) {
  int cout = threadIdx.x;
  float iscale = wsf[WS_ISCALE];
  float oscale = wsf[WS_OSCALE];
  wsf[WS_MFIXED + cout] = rintf(iscale * wsf[WS_WSCALE + cout] / oscale * 65536.f);
}

__global__ void k_requant(int* __restrict__ io, const float* __restrict__ wsf, int n) {
  int idx = blockIdx.x * blockDim.x + threadIdx.x;
  if (idx >= n) return;
  int cout = (idx / HW) & 255;
  float M = wsf[WS_MFIXED + cout];
  float av = (float)io[idx];
  float o = rintf(av * M / 65536.f);
  float lo = cout < 128 ? -8.f : -128.f;
  float hi = cout < 128 ? 7.f : 127.f;
  o = fminf(fmaxf(o, lo), hi);
  ((float*)io)[idx] = o;
}

extern "C" void kernel_launch(void* const* d_in, const int* in_sizes, int n_in,
                              void* d_out, int out_size, void* d_ws, size_t ws_size,
                              hipStream_t stream) {
  const float* x = (const float*)d_in[0];
  const float* w = (const float*)d_in[1];
  const float* b = (const float*)d_in[2];
  int* wsi = (int*)d_ws;
  float* wsf = (float*)d_ws;
  signed char* xq = (signed char*)d_ws + WS_XQ_OFF;
  signed char* w8 = (signed char*)d_ws + WS_W8_OFF;
  int nx = in_sizes[0];           // 6,422,528
  int nout = out_size;            // 12,845,056

  k_init<<<1, 32, 0, stream>>>(wsi);
  k_minmax<<<1024, 256, 0, stream>>>((const float4*)x, nx / 4, wsi);
  k_scale<<<1, 1, 0, stream>>>(wsi, wsf);
  k_wquant<<<COUT, 128, 0, stream>>>(w, b, wsi, wsf, w8);
  dim3 qgrid(HW / 32, CIN / 32, BATCH);
  k_xquant<<<qgrid, 256, 0, stream>>>(x, wsf, xq);
  dim3 grid(MTOT / 64, COUT / 64);
  k_conv<<<grid, 256, 0, stream>>>(xq, w8, wsi, (int*)d_out, &wsi[WS_ACCMAX0]);
  k_outscale<<<1, 1, 0, stream>>>(wsi, wsf);
  k_mfixed<<<1, 256, 0, stream>>>(wsf);
  k_requant<<<(nout + 255) / 256, 256, 0, stream>>>((int*)d_out, wsf, nout);
}
